// Attention_58093727646309
// MI455X (gfx1250) — matrix operand order repaired and checked by reading, not compiled, not run on hardware
//
#include <hip/hip_runtime.h>
#include <hip/hip_bf16.h>

// Sizes fixed by the reference
#define B_ 16
#define S_ 512
#define D_ 256
#define L_ 8192

typedef __attribute__((ext_vector_type(16))) __bf16        v16bf;
typedef __attribute__((ext_vector_type(8)))  __bf16        bf16x8;
typedef __attribute__((ext_vector_type(4)))  __bf16        bf16x4;
typedef __attribute__((ext_vector_type(8)))  float         v8f;
typedef __attribute__((ext_vector_type(4)))  float         f32x4;
typedef __attribute__((ext_vector_type(4)))  unsigned int  u32x4;

union FragBF { v16bf v; u32x4 q[2]; };

__device__ __forceinline__ void sched_fence() {
#if __has_builtin(__builtin_amdgcn_sched_barrier)
  __builtin_amdgcn_sched_barrier(0);   // nothing crosses: pin loads above WMMAs
#endif
}

// ---- JAX threefry2x32 for jax.random.bernoulli(key(42), 0.8, (B,S,L)) ----
__device__ __forceinline__ void tfround(unsigned &x0, unsigned &x1, int r) {
  x0 += x1; x1 = (x1 << r) | (x1 >> (32 - r)); x1 ^= x0;
}
__device__ __forceinline__ bool keep_flag(unsigned i) {
  const unsigned H = 1u << 25;               // B*S*L/2 (counter halves)
  const bool lo = i < H;
  unsigned c0 = lo ? i : i - H;
  unsigned c1 = lo ? i + H : i;
  const unsigned k0 = 0u, k1 = 42u, k2 = 0x1BD11BDAu ^ k0 ^ k1;
  unsigned x0 = c0 + k0, x1 = c1 + k1;
  tfround(x0,x1,13); tfround(x0,x1,15); tfround(x0,x1,26); tfround(x0,x1, 6);
  x0 += k1; x1 += k2 + 1u;
  tfround(x0,x1,17); tfround(x0,x1,29); tfround(x0,x1,16); tfround(x0,x1,24);
  x0 += k2; x1 += k0 + 2u;
  tfround(x0,x1,13); tfround(x0,x1,15); tfround(x0,x1,26); tfround(x0,x1, 6);
  x0 += k0; x1 += k1 + 3u;
  tfround(x0,x1,17); tfround(x0,x1,29); tfround(x0,x1,16); tfround(x0,x1,24);
  x0 += k1; x1 += k2 + 4u;
  tfround(x0,x1,13); tfround(x0,x1,15); tfround(x0,x1,26); tfround(x0,x1, 6);
  x0 += k2; x1 += k0 + 5u;
  unsigned bits = lo ? x0 : x1;
  float u = __uint_as_float((bits >> 9) | 0x3f800000u) - 1.0f;
  return u < 0.8f;                           // keep prob = 1 - DROP_P
}

__device__ __forceinline__ bf16x4 cvt4(f32x4 v) {
  bf16x4 h;                                  // lowers to v_cvt_pk_bf16_f32
  h[0] = (__bf16)v.x; h[1] = (__bf16)v.y; h[2] = (__bf16)v.z; h[3] = (__bf16)v.w;
  return h;
}

// ============ Pass 0: one-time f32 -> bf16 conversion of label_emb ===========
__global__ __launch_bounds__(256) void cvt_emb_bf16(
    const float* __restrict__ emb, __bf16* __restrict__ ebf) {
  size_t base = ((size_t)blockIdx.x * 256 + threadIdx.x) * 8;   // 8 f32 / thread
  f32x4 a = *reinterpret_cast<const f32x4*>(emb + base);
  f32x4 b = *reinterpret_cast<const f32x4*>(emb + base + 4);
  *reinterpret_cast<bf16x4*>(ebf + base)     = cvt4(a);
  *reinterpret_cast<bf16x4*>(ebf + base + 4) = cvt4(b);
}

// =====================  Pass 1: per-(b,s) softmax stats  =====================
// Grid (S/16, B), 256 threads (8 waves). Each wave scans 64 l-tiles with
// v_wmma_f32_16x16x32_bf16 and keeps online (max, sumexp) per score row.
// EBF=true: all 8 B-fragments preloaded (16 pipelined b128 loads) per l-tile,
// pinned above the WMMA chain with a sched_barrier.
template <bool EBF>
__global__ __launch_bounds__(256) void lblattn_rowstats(
    const float* __restrict__ x, const float* __restrict__ emb,
    const __bf16* __restrict__ ebf,
    float* __restrict__ mws, float* __restrict__ zws) {
  __shared__ __align__(16) __bf16 sX[16 * D_];           // x tile bf16
  __shared__ float wgm[8][16];
  __shared__ float wgz[8][16];
  const int tid = threadIdx.x, w = tid >> 5, lane = tid & 31;
  const int g = lane >> 4, ln = lane & 15;
  const int b = blockIdx.y, sbase = blockIdx.x * 16;

  for (int r = 0; r < 4; ++r) {              // 16x256 f32 -> bf16 LDS
    int idx4 = tid + r * 256;
    int e = idx4 * 4, s = e >> 8, d = e & 255;
    f32x4 v = *reinterpret_cast<const f32x4*>(x + ((size_t)b * S_ + sbase + s) * D_ + d);
    *reinterpret_cast<bf16x4*>(&sX[s * D_ + d]) = cvt4(v);
  }
  __syncthreads();

  FragBF A[8];                               // A = x[16s,32d] per K-step (contiguous k = 16h + s)
  for (int k = 0; k < 8; ++k) {
    int k0 = k * 32;
    A[k].q[0] = *reinterpret_cast<const u32x4*>(&sX[ln * D_ + k0 + g * 16]);
    A[k].q[1] = *reinterpret_cast<const u32x4*>(&sX[ln * D_ + k0 + g * 16 + 8]);
  }
  float mr[8], zr[8];
  for (int i = 0; i < 8; ++i) { mr[i] = -1e30f; zr[i] = 0.f; }
  const v8f vzero = {0.f,0.f,0.f,0.f,0.f,0.f,0.f,0.f};

  for (int lt = w * 64; lt < w * 64 + 64; ++lt) {
    v8f c = vzero;
    if (EBF) {                               // bf16 E: batch loads, then WMMA chain
      const __bf16* Er = ebf + ((size_t)b * L_ + lt * 16 + ln) * D_;
      FragBF Bf[8];
#pragma unroll
      for (int k = 0; k < 8; ++k) {          // 16 b128 loads, deep MLP
        int k0 = k * 32 + g * 16;
        Bf[k].q[0] = *reinterpret_cast<const u32x4*>(Er + k0);
        Bf[k].q[1] = *reinterpret_cast<const u32x4*>(Er + k0 + 8);
      }
      sched_fence();                         // keep all loads above the WMMAs
#pragma unroll
      for (int k = 0; k < 8; ++k)
        c = __builtin_amdgcn_wmma_f32_16x16x32_bf16(false, A[k].v, false, Bf[k].v,
                                                    (short)0, c, false, false);
    } else {                                 // fallback: convert f32 E on the fly
      const float* Er = emb + ((size_t)b * L_ + lt * 16 + ln) * D_;
      for (int k = 0; k < 8; ++k) {
        int k0 = k * 32 + g * 16;
        f32x4 e0 = *reinterpret_cast<const f32x4*>(Er + k0);
        f32x4 e1 = *reinterpret_cast<const f32x4*>(Er + k0 + 4);
        f32x4 e2 = *reinterpret_cast<const f32x4*>(Er + k0 + 8);
        f32x4 e3 = *reinterpret_cast<const f32x4*>(Er + k0 + 12);
        FragBF Bf;
        Bf.v[ 0]=(__bf16)e0.x; Bf.v[ 1]=(__bf16)e0.y; Bf.v[ 2]=(__bf16)e0.z; Bf.v[ 3]=(__bf16)e0.w;
        Bf.v[ 4]=(__bf16)e1.x; Bf.v[ 5]=(__bf16)e1.y; Bf.v[ 6]=(__bf16)e1.z; Bf.v[ 7]=(__bf16)e1.w;
        Bf.v[ 8]=(__bf16)e2.x; Bf.v[ 9]=(__bf16)e2.y; Bf.v[10]=(__bf16)e2.z; Bf.v[11]=(__bf16)e2.w;
        Bf.v[12]=(__bf16)e3.x; Bf.v[13]=(__bf16)e3.y; Bf.v[14]=(__bf16)e3.z; Bf.v[15]=(__bf16)e3.w;
        c = __builtin_amdgcn_wmma_f32_16x16x32_bf16(false, A[k].v, false, Bf.v,
                                                    (short)0, c, false, false);
      }
    }
    for (int i = 0; i < 8; ++i) {            // online softmax, row M = i + g*8
      float sc = c[i];
      float mn = fmaxf(mr[i], sc);
      zr[i] = zr[i] * __expf(mr[i] - mn) + __expf(sc - mn);
      mr[i] = mn;
    }
  }
  // reduce over the 16 lanes of each half (each lane held one l column)
  for (int off = 1; off < 16; off <<= 1) {
    for (int i = 0; i < 8; ++i) {
      float m2 = __shfl_xor(mr[i], off, 32);
      float z2 = __shfl_xor(zr[i], off, 32);
      float mn = fmaxf(mr[i], m2);
      zr[i] = zr[i] * __expf(mr[i] - mn) + z2 * __expf(m2 - mn);
      mr[i] = mn;
    }
  }
  if (ln == 0)
    for (int i = 0; i < 8; ++i) { wgm[w][g * 8 + i] = mr[i]; wgz[w][g * 8 + i] = zr[i]; }
  __syncthreads();
  if (tid < 16) {                            // combine 8 waves' partials
    float m = -1e30f;
    for (int ww = 0; ww < 8; ++ww) m = fmaxf(m, wgm[ww][tid]);
    float z = 0.f;
    for (int ww = 0; ww < 8; ++ww) z += wgz[ww][tid] * __expf(wgm[ww][tid] - m);
    mws[(size_t)b * S_ + sbase + tid] = m;
    zws[(size_t)b * S_ + sbase + tid] = z;
  }
}

// =====================  Pass 2: P = softmax/mask/drop, out = P^T x  ==========
// Grid (L/32, B), 256 threads. Per WG: 32 labels x 256 dims of output.
template <bool EBF>
__global__ __launch_bounds__(256) void lblattn_pool(
    const float* __restrict__ x, const float* __restrict__ emb,
    const __bf16* __restrict__ ebf,
    const unsigned char* __restrict__ mask,
    const float* __restrict__ mws, const float* __restrict__ zws,
    float* __restrict__ out) {
  __shared__ __align__(16) __bf16 sE [32 * D_];          // E tile  [l][d]
  __shared__ __align__(16) __bf16 sXl[32 * D_];          // x chunk [s][d]
  __shared__ __align__(16) __bf16 sXT[D_ * 32];          // x chunk [d][s]
  __shared__ __align__(16) __bf16 sP [32 * 32];          // P^T     [l][s]
  __shared__ float sM[32]; __shared__ float sR[32];      // row max, 1/Z
  __shared__ unsigned char sMk[32];
  const int tid = threadIdx.x, w = tid >> 5, lane = tid & 31;
  const int g = lane >> 4, ln = lane & 15;
  const int b = blockIdx.y, l0 = blockIdx.x * 32;
  const v8f vzero = {0.f,0.f,0.f,0.f,0.f,0.f,0.f,0.f};

  if (EBF) {                                 // E tile: straight bf16 copy
    for (int r = 0; r < 4; ++r) {
      int idx8 = tid + r * 256;
      int e = idx8 * 8, l = e >> 8, d = e & 255;
      *reinterpret_cast<u32x4*>(&sE[l * D_ + d]) =
          *reinterpret_cast<const u32x4*>(ebf + ((size_t)b * L_ + l0 + l) * D_ + d);
    }
  } else {                                   // E tile: f32 -> bf16
    for (int r = 0; r < 8; ++r) {
      int idx4 = tid + r * 256;
      int e = idx4 * 4, l = e >> 8, d = e & 255;
      f32x4 v = *reinterpret_cast<const f32x4*>(emb + ((size_t)b * L_ + l0 + l) * D_ + d);
      *reinterpret_cast<bf16x4*>(&sE[l * D_ + d]) = cvt4(v);
    }
  }
  v8f acc[4];
  for (int j = 0; j < 4; ++j) acc[j] = vzero;

  for (int s0 = 0; s0 < S_; s0 += 32) {
    __syncthreads();                         // protect LDS reuse across iters
    for (int r = 0; r < 8; ++r) {            // x chunk: row-major + transposed
      int idx4 = tid + r * 256;
      int e = idx4 * 4, s = e >> 8, d = e & 255;
      f32x4 v = *reinterpret_cast<const f32x4*>(x + ((size_t)b * S_ + s0 + s) * D_ + d);
      bf16x4 h = cvt4(v);
      *reinterpret_cast<bf16x4*>(&sXl[s * D_ + d]) = h;
      sXT[(d + 0) * 32 + s] = h[0];
      sXT[(d + 1) * 32 + s] = h[1];
      sXT[(d + 2) * 32 + s] = h[2];
      sXT[(d + 3) * 32 + s] = h[3];
    }
    if (tid < 32) {
      sM[tid]  = mws[(size_t)b * S_ + s0 + tid];
      sR[tid]  = 1.0f / zws[(size_t)b * S_ + s0 + tid];
      sMk[tid] = mask[(size_t)b * S_ + s0 + tid];
    }
    __syncthreads();

    if (w < 4) {                             // GEMM1: scores[32s,32l], 4 C-tiles
      const int ss = (w & 1) * 16, ll = (w >> 1) * 16;
      v8f c = vzero;
      for (int k = 0; k < 8; ++k) {
        int k0 = k * 32;
        FragBF Af, Bf;
        Af.q[0] = *reinterpret_cast<const u32x4*>(&sXl[(ss + ln) * D_ + k0 + g * 16]);
        Af.q[1] = *reinterpret_cast<const u32x4*>(&sXl[(ss + ln) * D_ + k0 + g * 16 + 8]);
        Bf.q[0] = *reinterpret_cast<const u32x4*>(&sE [(ll + ln) * D_ + k0 + g * 16]);
        Bf.q[1] = *reinterpret_cast<const u32x4*>(&sE [(ll + ln) * D_ + k0 + g * 16 + 8]);
        c = __builtin_amdgcn_wmma_f32_16x16x32_bf16(false, Af.v, false, Bf.v,
                                                    (short)0, c, false, false);
      }
      const int l = l0 + ll + ln;            // C: N = lane%16 (label), M = i + g*8 (s)
      bf16x8 pv;
      for (int i = 0; i < 8; ++i) {
        int srow = ss + i + g * 8;
        float p = __expf(c[i] - sM[srow]) * sR[srow];
        if (!sMk[srow]) p = 0.f;
        unsigned fi = (unsigned)(b * S_ + s0 + srow) * (unsigned)L_ + (unsigned)l;
        p = keep_flag(fi) ? p * 1.25f : 0.f; // 1/(1-DROP_P)
        pv[i] = (__bf16)p;
      }
      *reinterpret_cast<bf16x8*>(&sP[(ll + ln) * 32 + ss + g * 8]) = pv;  // [l][s]
    }
    __syncthreads();

    // GEMM2: acc[16l,64d] += P^T[16l,32s] * x[32s,64d]   (all 8 waves)
    const int l2 = (w & 1) * 16, db = (w >> 1) * 64;
    FragBF Ap;
    Ap.q[0] = *reinterpret_cast<const u32x4*>(&sP[(l2 + ln) * 32 + g * 16]);
    Ap.q[1] = *reinterpret_cast<const u32x4*>(&sP[(l2 + ln) * 32 + g * 16 + 8]);
    for (int j = 0; j < 4; ++j) {
      FragBF Bx;
      int d = db + j * 16 + ln;
      Bx.q[0] = *reinterpret_cast<const u32x4*>(&sXT[d * 32 + g * 16]);
      Bx.q[1] = *reinterpret_cast<const u32x4*>(&sXT[d * 32 + g * 16 + 8]);
      acc[j] = __builtin_amdgcn_wmma_f32_16x16x32_bf16(false, Ap.v, false, Bx.v,
                                                       (short)0, acc[j], false, false);
    }
  }

  const int l2 = (w & 1) * 16, db = (w >> 1) * 64;
  for (int j = 0; j < 4; ++j)
    for (int i = 0; i < 8; ++i) {
      int l = l0 + l2 + i + g * 8;
      int d = db + j * 16 + ln;
      out[((size_t)b * L_ + l) * D_ + d] = acc[j][i];
    }
}

extern "C" void kernel_launch(void* const* d_in, const int* in_sizes, int n_in,
                              void* d_out, int out_size, void* d_ws, size_t ws_size,
                              hipStream_t stream) {
  (void)in_sizes; (void)n_in; (void)out_size;
  const float*         x    = (const float*)d_in[0];
  const unsigned char* mask = (const unsigned char*)d_in[1];  // jnp bool -> 1 byte
  const float*         emb  = (const float*)d_in[2];
  float* out = (float*)d_out;

  // Workspace layout: [ m (B*S f32) | Z (B*S f32) | optional bf16 E (B*L*D) ]
  float*  mws = (float*)d_ws;
  float*  zws = mws + (size_t)B_ * S_;
  const size_t statBytes = (size_t)2 * B_ * S_ * sizeof(float);      // 64 KB
  const size_t ebfBytes  = (size_t)B_ * L_ * D_ * sizeof(__bf16);    // 64 MB
  __bf16* ebf = (__bf16*)((char*)d_ws + statBytes);
  const bool useBf = ws_size >= statBytes + ebfBytes;

  if (useBf) {
    const size_t nElem = (size_t)B_ * L_ * D_;                       // 2^25
    cvt_emb_bf16<<<dim3((unsigned)(nElem / (256 * 8))), 256, 0, stream>>>(emb, ebf);
    lblattn_rowstats<true><<<dim3(S_ / 16, B_), 256, 0, stream>>>(x, emb, ebf, mws, zws);
    lblattn_pool<true><<<dim3(L_ / 32, B_), 256, 0, stream>>>(x, emb, ebf, mask, mws, zws, out);
  } else {
    lblattn_rowstats<false><<<dim3(S_ / 16, B_), 256, 0, stream>>>(x, emb, nullptr, mws, zws);
    lblattn_pool<false><<<dim3(L_ / 32, B_), 256, 0, stream>>>(x, emb, nullptr, mask, mws, zws, out);
  }
}
